// Attention_56118042690099
// MI455X (gfx1250) — matrix operand order repaired and checked by reading, not compiled, not run on hardware
//
#include <hip/hip_runtime.h>
#include <hip/hip_bf16.h>

typedef __bf16 bf16;
typedef __attribute__((ext_vector_type(16))) __bf16 v16bf;
typedef __attribute__((ext_vector_type(8)))  float  v8f;

constexpr int cB   = 4;
constexpr int cS   = 1024;
constexpr int cD   = 4096;
constexpr int cHQ  = 32;
constexpr int cHKV = 8;
constexpr int cHD  = 128;
constexpr int cM   = cB * cS;       // 4096 rows for all GEMMs

#ifndef __has_builtin
#define __has_builtin(x) 0
#endif
#if __has_builtin(__builtin_amdgcn_tensor_load_to_lds) && \
    __has_builtin(__builtin_amdgcn_s_wait_tensorcnt)
#define USE_TDM 1
#else
#define USE_TDM 0
#endif

static __device__ __forceinline__ v8f wmma_bf16(v16bf a, v16bf b, v8f c) {
  // D = A(16x32 bf16) * B(32x16 bf16) + C(16x16 f32)
  return __builtin_amdgcn_wmma_f32_16x16x32_bf16(false, a, false, b, (short)0, c,
                                                 false, false);
}

union FragU { v16bf v; unsigned u[8]; uint4 q[2]; };

// B/K/V-style fragment: 16 contiguous bf16 (lane-half picks K range outside).
static __device__ __forceinline__ v16bf ld_frag32(const bf16* p) {
  FragU f;
  f.q[0] = *(const uint4*)p;
  f.q[1] = *(const uint4*)(p + 8);
  return f.v;
}
// A/P/Q-style fragment: two 8-elem chunks at +0 and +16 elements.
static __device__ __forceinline__ v16bf ld_fragA(const bf16* p) {
  FragU f;
  f.q[0] = *(const uint4*)p;
  f.q[1] = *(const uint4*)(p + 16);
  return f.v;
}

#if USE_TDM
typedef __attribute__((ext_vector_type(4))) unsigned tdm_u4;
typedef __attribute__((ext_vector_type(8))) int      tdm_i8;
typedef __attribute__((ext_vector_type(4))) int      tdm_i4;

// Issue one 2D TDM tile load (bf16 elements): global [d1][d0] -> LDS packed.
static __device__ __forceinline__ void tdm_load_2d(
    unsigned lds_off, const void* gaddr, unsigned tensor_d0, unsigned tensor_d1,
    unsigned tile_d0, unsigned tile_d1, unsigned d0_stride) {
  unsigned long ga = (unsigned long)gaddr;
  tdm_u4 g0;
  g0[0] = 1u;                                    // count=1, user descriptor
  g0[1] = lds_off;                               // lds_addr
  g0[2] = (unsigned)ga;                          // global_addr[31:0]
  g0[3] = (unsigned)((ga >> 32) & 0x1FFFFFFu) | (2u << 30);  // addr hi | type=2
  tdm_i8 g1;
  g1[0] = (int)(1u << 16);                       // wg_mask=0, data_size=1 (2B)
  g1[1] = (int)((tensor_d0 & 0xFFFFu) << 16);    // tensor_dim0 lo16 @bits63:48
  g1[2] = (int)(((tensor_d0 >> 16) & 0xFFFFu) | ((tensor_d1 & 0xFFFFu) << 16));
  g1[3] = (int)(((tensor_d1 >> 16) & 0xFFFFu) | ((tile_d0 & 0xFFFFu) << 16));
  g1[4] = (int)(tile_d1 & 0xFFFFu);              // tile_dim1; tile_dim2=0
  g1[5] = (int)d0_stride;                        // tensor_dim0_stride[31:0]
  g1[6] = 0;                                     // stride hi, dim1_stride lo
  g1[7] = 0;
  tdm_i4 z4 = {0, 0, 0, 0};
#if defined(__clang_major__) && (__clang_major__ >= 23)
  tdm_i8 z8 = {0, 0, 0, 0, 0, 0, 0, 0};
  __builtin_amdgcn_tensor_load_to_lds(g0, g1, z4, z4, z8, 0);
#else
  __builtin_amdgcn_tensor_load_to_lds(g0, g1, z4, z4, 0);
#endif
}
#endif  // USE_TDM

// ---------------------------------------------------------------- convert ---
__global__ void cvt_f32_bf16(const float* __restrict__ s, bf16* __restrict__ d,
                             long n) {
  long i = (long)blockIdx.x * blockDim.x + threadIdx.x;
  long stride = (long)gridDim.x * blockDim.x;
  for (; i < n; i += stride) d[i] = (bf16)s[i];
}

// Tiled transpose + convert: src f32 [K,N] -> dst bf16 [N,K].
__global__ void __launch_bounds__(256)
cvt_tr_bf16(const float* __restrict__ src, bf16* __restrict__ dst, int K,
            int N) {
  __shared__ float t[32][33];
  const int kb = blockIdx.y * 32, nb = blockIdx.x * 32;
  const int tx = threadIdx.x & 31, ty = threadIdx.x >> 5;  // 32x8
  for (int i = 0; i < 32; i += 8)
    t[ty + i][tx] = src[(long)(kb + ty + i) * N + nb + tx];
  __syncthreads();
  for (int i = 0; i < 32; i += 8)
    dst[(long)(nb + ty + i) * K + kb + tx] = (bf16)t[tx][ty + i];
}

// ------------------------------------------------------------------- GEMM ---
// C[M,N] (f32) = A[M,K](bf16,row-major) @ Wt[N,K](bf16, pre-transposed) + bias
// Block: 256 threads = 8 waves.  Tile 128x128, K-stage 64 (two WMMA K-steps).
// TDM path: double-buffered LDS, wave 0 DMAs stage i+1 while all compute i.
static __device__ __forceinline__ void gemm_stage_compute(
    const bf16* As, const bf16* Bs, int wr, int wc, int l16, int half,
    v8f (&acc)[4][2]) {
  for (int c = 0; c < 2; ++c) {
    v16bf af[4], bfr[2];
    for (int i = 0; i < 4; ++i)
      af[i] = ld_fragA(&As[(wr + i * 16 + l16) * 64 + c * 32 + half * 8]);
    for (int j = 0; j < 2; ++j)
      bfr[j] = ld_fragA(&Bs[(wc + j * 16 + l16) * 64 + c * 32 + half * 8]);
    for (int i = 0; i < 4; ++i)
      for (int j = 0; j < 2; ++j)
        acc[i][j] = wmma_bf16(af[i], bfr[j], acc[i][j]);
  }
}

__global__ void __launch_bounds__(256)
gemm_bf16(const bf16* __restrict__ A, const bf16* __restrict__ Wt,
          const float* __restrict__ bias, float* __restrict__ C,
          int M, int N, int K) {
  __shared__ bf16 As[2][128 * 64];   // [row][k]  16 KB x2
  __shared__ bf16 Bs[2][128 * 64];   // [col][k]  16 KB x2

  const int tid  = threadIdx.x;
  const int lane = tid & 31;
  const int wave = tid >> 5;
  const int half = lane >> 4;
  const int l16  = lane & 15;
  const int rowBase = blockIdx.y * 128;
  const int colBase = blockIdx.x * 128;
  const int wr = (wave >> 2) * 64;
  const int wc = (wave & 3) * 32;

  const v8f zero8 = {0.f, 0.f, 0.f, 0.f, 0.f, 0.f, 0.f, 0.f};
  v8f acc[4][2];
  for (int i = 0; i < 4; ++i)
    for (int j = 0; j < 2; ++j) acc[i][j] = zero8;

  const bf16* Ap = A + (long)rowBase * K;
  const bf16* Bp = Wt + (long)colBase * K;
  const int niter = K / 64;

#if USE_TDM
  if (wave == 0) {
    tdm_load_2d((unsigned)(unsigned long)&As[0][0], Ap, K, M, 64, 128, K);
    tdm_load_2d((unsigned)(unsigned long)&Bs[0][0], Bp, K, N, 64, 128, K);
  }
  for (int it = 0; it < niter; ++it) {
    const int cur = it & 1;
    if (wave == 0) {
      if (it + 1 < niter) {
        tdm_load_2d((unsigned)(unsigned long)&As[cur ^ 1][0],
                    Ap + (it + 1) * 64, K, M, 64, 128, K);
        tdm_load_2d((unsigned)(unsigned long)&Bs[cur ^ 1][0],
                    Bp + (it + 1) * 64, K, N, 64, 128, K);
        __builtin_amdgcn_s_wait_tensorcnt(2);  // stage `it` tiles landed
      } else {
        __builtin_amdgcn_s_wait_tensorcnt(0);
      }
    }
    __syncthreads();
    gemm_stage_compute(As[cur], Bs[cur], wr, wc, l16, half, acc);
    __syncthreads();  // buffers free before next overwrite
  }
#else
  for (int it = 0; it < niter; ++it) {
    int k0 = it * 64;
    for (int i = 0; i < 8; ++i) {
      int e = (tid + i * 256) * 4;            // 0..8191, multiple of 4
      int r = e >> 6, c = e & 63;
      *(uint2*)&As[0][e] = *(const uint2*)&Ap[(long)r * K + k0 + c];
      *(uint2*)&Bs[0][e] = *(const uint2*)&Bp[(long)r * K + k0 + c];
    }
    __syncthreads();
    gemm_stage_compute(As[0], Bs[0], wr, wc, l16, half, acc);
    __syncthreads();
  }
#endif

  for (int i = 0; i < 4; ++i)
    for (int j = 0; j < 2; ++j) {
      int col = colBase + wc + j * 16 + l16;
      float bv = bias ? bias[col] : 0.f;
      for (int r = 0; r < 8; ++r) {
        int row = rowBase + wr + i * 16 + r + 8 * half;
        C[(long)row * N + col] = acc[i][j][r] + bv;
      }
    }
}

// ---------------------------------------------------------------- rotary ----
// src f32 [B,S,H,HD] -> dst bf16 [B,H,S,HD], with rotary applied.
__global__ void rope_transpose(const float* __restrict__ src,
                               bf16* __restrict__ dst,
                               const float* __restrict__ fc,
                               const int* __restrict__ sp_ptr, int H) {
  const int sp = sp_ptr[0];
  long total = (long)cB * cS * H * (cHD / 2);
  long i = (long)blockIdx.x * blockDim.x + threadIdx.x;
  long stride = (long)gridDim.x * blockDim.x;
  for (; i < total; i += stride) {
    int d2 = (int)(i % (cHD / 2));
    long t = i / (cHD / 2);
    int h = (int)(t % H); t /= H;
    int s = (int)(t % cS);
    int b = (int)(t / cS);
    const float* p = src + (((long)(b * cS + s) * H + h) * cHD);
    float a  = p[2 * d2];
    float bb = p[2 * d2 + 1];
    const float* f = fc + ((long)(s + sp) * (cHD / 2) + d2) * 2;
    float c = f[0], d = f[1];
    bf16* q = dst + (((long)(b * H + h) * cS + s) * cHD);
    q[2 * d2]     = (bf16)(a * c - bb * d);
    q[2 * d2 + 1] = (bf16)(a * d + bb * c);
  }
}

// V: src f32 [B,S,H,HD] -> dst bf16 [B,H,HD,S] (dim-major for b128 B-frags)
__global__ void transpose_v(const float* __restrict__ src,
                            bf16* __restrict__ dst, int H) {
  long total = (long)cB * cS * H * cHD;
  long i = (long)blockIdx.x * blockDim.x + threadIdx.x;
  long stride = (long)gridDim.x * blockDim.x;
  for (; i < total; i += stride) {
    int d = (int)(i % cHD);
    long t = i / cHD;
    int h = (int)(t % H); t /= H;
    int s = (int)(t % cS);
    int b = (int)(t / cS);
    dst[((long)(b * H + h) * cHD + d) * cS + s] =
        (bf16)src[(((long)(b * cS + s) * H + h) * cHD) + d];
  }
}

// --------------------------------------------------------- flash attention --
// Q bf16 [B,HQ,S,HD], K bf16 [B,HKV,S,HD], V bf16 [B,HKV,HD,S],
// out bf16 [B,S,HQ*HD].  grid = (B*HQ, S/128); block = 256.
// Dynamic LDS: 2 x (Ks 32KB + Vs 32KB) = 128 KB (double-buffered for TDM).
__global__ void __launch_bounds__(256)
flash_attn(const bf16* __restrict__ Q, const bf16* __restrict__ Kg,
           const bf16* __restrict__ Vg, bf16* __restrict__ O) {
  extern __shared__ bf16 smem[];   // [buf][Ks 16384 | Vs 16384]

  const int b  = blockIdx.x / cHQ;
  const int h  = blockIdx.x % cHQ;
  const int hk = h / (cHQ / cHKV);
  const int qbase = blockIdx.y * 128;

  const int tid  = threadIdx.x;
  const int lane = tid & 31;
  const int wave = tid >> 5;
  const int half = lane >> 4;
  const int l16  = lane & 15;
  const float sm_scale = 0.08838834764831845f;  // 1/sqrt(128)

  // Q fragments: 16 rows x HD=128 -> 4 A-fragments, straight from global.
  v16bf qf[4];
  {
    int row = qbase + wave * 16 + l16;
    const bf16* qp = Q + ((long)(b * cHQ + h) * cS + row) * cHD;
    for (int c = 0; c < 4; ++c) qf[c] = ld_fragA(qp + c * 32 + half * 8);
  }

  const v8f zero8 = {0.f, 0.f, 0.f, 0.f, 0.f, 0.f, 0.f, 0.f};
  float m_s[8], l_s[8];
  v8f o_acc[8];
  for (int r = 0; r < 8; ++r) { m_s[r] = -1e30f; l_s[r] = 0.f; }
  for (int n = 0; n < 8; ++n) o_acc[n] = zero8;

  const bf16* kp = Kg + (long)(b * cHKV + hk) * cS * cHD;   // [S][HD]
  const bf16* vp = Vg + (long)(b * cHKV + hk) * cHD * cS;   // [HD][S]

  const int nkb = qbase / 128 + 1;   // causal
#if USE_TDM
  if (wave == 0) {
    tdm_load_2d((unsigned)(unsigned long)&smem[0], kp, cHD, cS, 128, 128, cHD);
    tdm_load_2d((unsigned)(unsigned long)&smem[16384], vp, cS, cHD, 128, 128,
                cS);
  }
#endif
  for (int kb = 0; kb < nkb; ++kb) {
#if USE_TDM
    const int cur = kb & 1;
    bf16* Ks = smem + cur * 32768;
    bf16* Vs = Ks + 16384;
    if (wave == 0) {
      if (kb + 1 < nkb) {
        bf16* Kn = smem + (cur ^ 1) * 32768;
        tdm_load_2d((unsigned)(unsigned long)Kn,
                    kp + (long)(kb + 1) * 128 * cHD, cHD, cS, 128, 128, cHD);
        tdm_load_2d((unsigned)(unsigned long)(Kn + 16384),
                    vp + (kb + 1) * 128, cS, cHD, 128, 128, cS);
        __builtin_amdgcn_s_wait_tensorcnt(2);   // block kb's tiles landed
      } else {
        __builtin_amdgcn_s_wait_tensorcnt(0);
      }
    }
    __syncthreads();
#else
    bf16* Ks = smem;
    bf16* Vs = smem + 16384;
    for (int i = 0; i < 16; ++i) {
      int e = (tid + i * 256) * 4;
      int r = e >> 7, c = e & 127;
      *(uint2*)&Ks[e] = *(const uint2*)&kp[(long)(kb * 128 + r) * cHD + c];
      *(uint2*)&Vs[e] = *(const uint2*)&vp[(long)r * cS + kb * 128 + c];
    }
    __syncthreads();
#endif

    // scores S = Q @ K^T : 16 rows x 128 keys, 8 sub-tiles
    v8f s_acc[8];
    for (int nb = 0; nb < 8; ++nb) s_acc[nb] = zero8;
    for (int nb = 0; nb < 8; ++nb) {
      int key = nb * 16 + l16;
      for (int c = 0; c < 4; ++c) {
        v16bf kf = ld_fragA(&Ks[key * 128 + c * 32 + half * 8]);
        s_acc[nb] = wmma_bf16(qf[c], kf, s_acc[nb]);
      }
    }
    __syncthreads();   // all waves done reading Ks before P overwrites it

    // online softmax; write P (bf16) into Ks (this wave's 16 rows)
    for (int r = 0; r < 8; ++r) {
      int prow = wave * 16 + r + 8 * half;
      int qg = qbase + prow;
      float rmax = -1e30f;
      for (int nb = 0; nb < 8; ++nb) {
        int kg = kb * 128 + nb * 16 + l16;
        float sv = (kg <= qg) ? s_acc[nb][r] * sm_scale : -1e30f;
        s_acc[nb][r] = sv;
        rmax = fmaxf(rmax, sv);
      }
      for (int m = 1; m < 16; m <<= 1)
        rmax = fmaxf(rmax, __shfl_xor(rmax, m, 32));
      float mnew = fmaxf(m_s[r], rmax);
      float alpha = __expf(m_s[r] - mnew);
      m_s[r] = mnew;
      float rsum = 0.f;
      for (int nb = 0; nb < 8; ++nb) {
        float pv = __expf(s_acc[nb][r] - mnew);
        rsum += pv;
        Ks[prow * 128 + nb * 16 + l16] = (bf16)pv;
      }
      for (int m = 1; m < 16; m <<= 1) rsum += __shfl_xor(rsum, m, 32);
      l_s[r] = l_s[r] * alpha + rsum;
      for (int n = 0; n < 8; ++n) o_acc[n][r] = o_acc[n][r] * alpha;
    }
    __syncthreads();   // P visible in LDS

    // O += P @ V
    for (int c = 0; c < 4; ++c) {
      v16bf pf = ld_fragA(&Ks[(wave * 16 + l16) * 128 + c * 32 + half * 8]);
      for (int n = 0; n < 8; ++n) {
        v16bf vf = ld_fragA(&Vs[(n * 16 + l16) * 128 + c * 32 + half * 8]);
        o_acc[n] = wmma_bf16(pf, vf, o_acc[n]);
      }
    }
    __syncthreads();   // buffer free before next DMA/copy overwrites it
  }

  for (int r = 0; r < 8; ++r) {
    int row = qbase + wave * 16 + r + 8 * half;
    float inv = 1.0f / l_s[r];
    bf16* op = O + ((long)(b * cS + row) * cD + h * cHD);
    for (int n = 0; n < 8; ++n)
      op[n * 16 + l16] = (bf16)(o_acc[n][r] * inv);
  }
}

// ------------------------------------------------------------------ launch --
extern "C" void kernel_launch(void* const* d_in, const int* in_sizes, int n_in,
                              void* d_out, int out_size, void* d_ws,
                              size_t ws_size, hipStream_t stream) {
  const float* x  = (const float*)d_in[0];
  const float* fc = (const float*)d_in[1];
  const float* Wq = (const float*)d_in[2];
  const float* bq = (const float*)d_in[3];
  const float* Wk = (const float*)d_in[4];
  const float* bk = (const float*)d_in[5];
  const float* Wv = (const float*)d_in[6];
  const float* bv = (const float*)d_in[7];
  const float* Wo = (const float*)d_in[8];
  const float* bo = (const float*)d_in[9];
  const int* startpos = (const int*)d_in[10];
  float* out = (float*)d_out;

  char* w = (char*)d_ws;
  auto alloc = [&](size_t bytes) -> char* {
    char* p = w;
    w += (bytes + 255) & ~(size_t)255;
    return p;
  };
  const size_t nX  = (size_t)cM * cD;
  const size_t nKV = (size_t)cM * (cHKV * cHD);

  bf16* xb  = (bf16*)alloc(nX * 2);
  bf16* Wqt = (bf16*)alloc((size_t)cD * cD * 2);              // [N=D][K=D]
  bf16* Wkt = (bf16*)alloc((size_t)cD * (cHKV * cHD) * 2);    // [1024][4096]
  bf16* Wvt = (bf16*)alloc((size_t)cD * (cHKV * cHD) * 2);
  bf16* Wot = (bf16*)alloc((size_t)cD * cD * 2);
  float* qf = (float*)alloc(nX * 4);
  float* kf = (float*)alloc(nKV * 4);
  float* vf = (float*)alloc(nKV * 4);
  bf16* qbr = (bf16*)alloc(nX * 2);
  bf16* kbr = (bf16*)alloc(nKV * 2);
  bf16* vbr = (bf16*)alloc(nKV * 2);
  bf16* ab  = (bf16*)alloc(nX * 2);

  const int CB = 2048;
  cvt_f32_bf16<<<CB, 256, 0, stream>>>(x, xb, (long)nX);
  dim3 tq(cD / 32, cD / 32), tkv((cHKV * cHD) / 32, cD / 32);
  cvt_tr_bf16<<<tq, 256, 0, stream>>>(Wq, Wqt, cD, cD);
  cvt_tr_bf16<<<tkv, 256, 0, stream>>>(Wk, Wkt, cD, cHKV * cHD);
  cvt_tr_bf16<<<tkv, 256, 0, stream>>>(Wv, Wvt, cD, cHKV * cHD);
  cvt_tr_bf16<<<tq, 256, 0, stream>>>(Wo, Wot, cD, cD);

  dim3 blk(256);
  dim3 gq(cD / 128, cM / 128);
  dim3 gkv((cHKV * cHD) / 128, cM / 128);
  gemm_bf16<<<gq, blk, 0, stream>>>(xb, Wqt, bq, qf, cM, cD, cD);
  gemm_bf16<<<gkv, blk, 0, stream>>>(xb, Wkt, bk, kf, cM, cHKV * cHD, cD);
  gemm_bf16<<<gkv, blk, 0, stream>>>(xb, Wvt, bv, vf, cM, cHKV * cHD, cD);

  rope_transpose<<<CB, 256, 0, stream>>>(qf, qbr, fc, startpos, cHQ);
  rope_transpose<<<CB, 256, 0, stream>>>(kf, kbr, fc, startpos, cHKV);
  transpose_v<<<CB, 256, 0, stream>>>(vf, vbr, cHKV);

  dim3 ga(cB * cHQ, cS / 128);
  // 128 KB dynamic LDS (double-buffered K/V tiles); fallback path uses half.
  flash_attn<<<ga, blk, 131072, stream>>>(qbr, kbr, vbr, ab);

  gemm_bf16<<<gq, blk, 0, stream>>>(ab, Wot, bo, out, cM, cD, cD);
}
